// GraphSageEncoder_59219009077768
// MI455X (gfx1250) — matrix operand order repaired and checked by reading, not compiled, not run on hardware
//
#include <hip/hip_runtime.h>

#define N_NODES 100000
#define N_EDGES 1600000
#define D_IN    128
#define D_HID   128
#define D_OUT   64

typedef __attribute__((ext_vector_type(16))) __bf16   v16bf;
typedef __attribute__((ext_vector_type(8)))  float    v8f;

// ---------- helpers ----------
// hardware f32 -> bf16 (lowers to v_cvt on gfx1250)
__device__ __forceinline__ unsigned short f2bf(float f) {
    return __builtin_bit_cast(unsigned short, (__bf16)f);
}
__device__ __forceinline__ unsigned int pk2bf(float lo, float hi) {
    return (unsigned int)f2bf(lo) | ((unsigned int)f2bf(hi) << 16);
}

// A operand (16x32 bf16, M x K) from LDS row-major [16][128] bf16.
// lane L: M = L%16; elem i: K = kbase + (i/8)*16 + (L/16)*8 + (i%8)
__device__ __forceinline__ v16bf load_a_frag(const unsigned short* lds, int row,
                                             int kbase, int nhalf) {
    const unsigned short* base = lds + row * 128 + kbase + nhalf * 8;
    union { uint4 q[2]; v16bf v; } U;
    U.q[0] = *reinterpret_cast<const uint4*>(base);
    U.q[1] = *reinterpret_cast<const uint4*>(base + 16);
    return U.v;
}

// B operand pre-swizzled in workspace: Wbf[(frag*32 + lane)*16 + i]
__device__ __forceinline__ v16bf load_b_pre(const unsigned short* __restrict__ Wbf,
                                            int frag, int lane) {
    const uint4* p = reinterpret_cast<const uint4*>(Wbf + ((long)frag * 32 + lane) * 16);
    union { uint4 q[2]; v16bf v; } U;
    U.q[0] = p[0];
    U.q[1] = p[1];
    return U.v;
}

// ---------- kernels ----------
__global__ void zero4_kernel(float4* __restrict__ p, long n4) {
    long i = (long)blockIdx.x * blockDim.x + threadIdx.x;
    long stride = (long)gridDim.x * blockDim.x;
    float4 z = {0.f, 0.f, 0.f, 0.f};
    for (; i < n4; i += stride) p[i] = z;
}

// Convert f32 weight matrix W[DOUT][128] into B-fragment-swizzled bf16.
// For element (j,k): frag = (j/16)*4 + (k/32); lane = ((k%16)/8)*16 + j%16; i = ((k%32)/16)*8 + k%8
__global__ void weight_prep_kernel(const float* __restrict__ W,
                                   unsigned short* __restrict__ Wbf, int dout) {
    int t = blockIdx.x * blockDim.x + threadIdx.x;
    if (t >= dout * 128) return;
    int j = t >> 7;
    int k = t & 127;
    int frag = (j >> 4) * 4 + (k >> 5);
    int lane = (((k >> 3) & 1) << 4) | (j & 15);
    int i    = (((k >> 4) & 1) << 3) | (k & 7);
    Wbf[((long)frag * 32 + lane) * 16 + i] = f2bf(W[t]);
}

__global__ void degree_kernel(const long long* __restrict__ edge,
                              float* __restrict__ deg) {
    int e = blockIdx.x * blockDim.x + threadIdx.x;
    if (e < N_EDGES) {
        int dst = (int)edge[N_EDGES + e];
        atomicAdd(&deg[dst], 1.0f);
    }
}

// one wave per edge; lane handles 4 consecutive floats of the 128-wide row
__global__ void scatter_kernel(const float* __restrict__ feat,
                               const long long* __restrict__ edge,
                               float* __restrict__ agg) {
    int tid  = blockIdx.x * blockDim.x + threadIdx.x;
    int e    = tid >> 5;
    int lane = tid & 31;
    if (e >= N_EDGES) return;
    int src = (int)edge[e];
    int dst = (int)edge[N_EDGES + e];
    const float4 v = *reinterpret_cast<const float4*>(feat + (long)src * 128 + lane * 4);
    float* d = agg + (long)dst * 128 + lane * 4;
    atomicAdd(d + 0, v.x);
    atomicAdd(d + 1, v.y);
    atomicAdd(d + 2, v.z);
    atomicAdd(d + 3, v.w);
}

// Fused SAGE linear: out[16 x DOUT] = (agg/deg) @ Wl^T + x @ Wr^T + b  (K=128)
// block: (DOUT/16) waves; wave w computes column tile [w*16, w*16+16)
template <int DOUT, bool RELU>
__global__ void sage_gemm_kernel(const float* __restrict__ agg,
                                 const float* __restrict__ deg,
                                 const float* __restrict__ x,
                                 const unsigned short* __restrict__ WlBf,
                                 const float* __restrict__ bias,
                                 const unsigned short* __restrict__ WrBf,
                                 float* __restrict__ out) {
    constexpr int K = 128;
    __shared__ __align__(16) unsigned short sA[16 * K];  // bf16(agg * 1/deg)
    __shared__ __align__(16) unsigned short sX[16 * K];  // bf16(x)
    __shared__ float sInv[16];

    const int tid     = threadIdx.x;
    const int rowBase = blockIdx.x * 16;

    if (tid < 16) {
        float d = deg[rowBase + tid];
        sInv[tid] = 1.0f / fmaxf(d, 1.0f);
    }
    __syncthreads();

    // stage activations as bf16 pairs (hardware cvt, packed dword stores)
    const int T = blockDim.x;
    const float2* agg2 = reinterpret_cast<const float2*>(agg);
    const float2* x2   = reinterpret_cast<const float2*>(x);
    unsigned int* sA32 = reinterpret_cast<unsigned int*>(sA);
    unsigned int* sX32 = reinterpret_cast<unsigned int*>(sX);
    for (int idx = tid; idx < 16 * (K / 2); idx += T) {
        int r  = idx >> 6;        // row in tile
        int c2 = idx & 63;        // float2 index along K
        long off = (long)(rowBase + r) * (K / 2) + c2;
        float2 a  = agg2[off];
        float2 xx = x2[off];
        float inv = sInv[r];
        sA32[idx] = pk2bf(a.x * inv, a.y * inv);
        sX32[idx] = pk2bf(xx.x, xx.y);
    }
    __syncthreads();

    const int wave  = tid >> 5;   // column tile
    const int lane  = tid & 31;
    const int nhalf = lane >> 4;
    const int nmod  = lane & 15;
    const int j     = wave * 16 + nmod;

    v8f acc = {};
#pragma unroll
    for (int kb = 0; kb < K; kb += 32) {
        const int frag = wave * (K / 32) + (kb >> 5);
        v16bf aAgg = load_a_frag(sA, nmod, kb, nhalf);
        v16bf aX   = load_a_frag(sX, nmod, kb, nhalf);
        v16bf bL   = load_b_pre(WlBf, frag, lane);
        v16bf bR   = load_b_pre(WrBf, frag, lane);
        acc = __builtin_amdgcn_wmma_f32_16x16x32_bf16(false, aAgg, false, bL,
                                                      (short)0, acc, false, false);
        acc = __builtin_amdgcn_wmma_f32_16x16x32_bf16(false, aX, false, bR,
                                                      (short)0, acc, false, false);
    }

    const float bj = bias[j];
#pragma unroll
    for (int r = 0; r < 8; ++r) {
        float v = acc[r] + bj;
        if (RELU) v = fmaxf(v, 0.0f);
        out[(long)(rowBase + r + nhalf * 8) * DOUT + j] = v;
    }
}

// ---------- launcher ----------
extern "C" void kernel_launch(void* const* d_in, const int* in_sizes, int n_in,
                              void* d_out, int out_size, void* d_ws, size_t ws_size,
                              hipStream_t stream) {
    const float*     x    = (const float*)d_in[0];       // [N,128]
    const long long* edge = (const long long*)d_in[1];   // [2,E] int64
    const float*     W_l1 = (const float*)d_in[2];       // [128,128]
    const float*     b1   = (const float*)d_in[3];       // [128]
    const float*     W_r1 = (const float*)d_in[4];       // [128,128]
    const float*     W_l2 = (const float*)d_in[5];       // [64,128]
    const float*     b2   = (const float*)d_in[6];       // [64]
    const float*     W_r2 = (const float*)d_in[7];       // [64,128]
    float*           out  = (float*)d_out;               // [N,64]

    float* ws  = (float*)d_ws;
    float* deg = ws;                                  // N
    float* agg = ws + N_NODES;                        // N*128
    float* h   = agg + (long)N_NODES * D_HID;         // N*128
    unsigned short* wbf = (unsigned short*)(h + (long)N_NODES * D_HID);
    unsigned short* Wl1Bf = wbf;                      // 128*128 bf16
    unsigned short* Wr1Bf = Wl1Bf + D_HID * D_IN;
    unsigned short* Wl2Bf = Wr1Bf + D_HID * D_IN;     // 64*128 bf16
    unsigned short* Wr2Bf = Wl2Bf + D_OUT * D_HID;

    const int  Z = 256;
    const long nAgg4 = (long)N_NODES * D_HID / 4;

    // weight pre-swizzle (tiny, L2-resident afterwards)
    weight_prep_kernel<<<(D_HID * D_IN + Z - 1) / Z, Z, 0, stream>>>(W_l1, Wl1Bf, D_HID);
    weight_prep_kernel<<<(D_HID * D_IN + Z - 1) / Z, Z, 0, stream>>>(W_r1, Wr1Bf, D_HID);
    weight_prep_kernel<<<(D_OUT * D_HID + Z - 1) / Z, Z, 0, stream>>>(W_l2, Wl2Bf, D_OUT);
    weight_prep_kernel<<<(D_OUT * D_HID + Z - 1) / Z, Z, 0, stream>>>(W_r2, Wr2Bf, D_OUT);

    // degrees (shared by both layers) + zero agg
    zero4_kernel<<<1024, Z, 0, stream>>>((float4*)deg, N_NODES / 4);
    zero4_kernel<<<4096, Z, 0, stream>>>((float4*)agg, nAgg4);
    degree_kernel<<<(N_EDGES + Z - 1) / Z, Z, 0, stream>>>(edge, deg);

    // layer 1: aggregate x, then fused GEMM + ReLU -> h
    scatter_kernel<<<(N_EDGES * 32 + Z - 1) / Z, Z, 0, stream>>>(x, edge, agg);
    sage_gemm_kernel<D_HID, true><<<N_NODES / 16, (D_HID / 16) * 32, 0, stream>>>(
        agg, deg, x, Wl1Bf, b1, Wr1Bf, h);

    // layer 2: aggregate h, fused GEMM -> out
    zero4_kernel<<<4096, Z, 0, stream>>>((float4*)agg, nAgg4);
    scatter_kernel<<<(N_EDGES * 32 + Z - 1) / Z, Z, 0, stream>>>(h, edge, agg);
    sage_gemm_kernel<D_OUT, false><<<N_NODES / 16, (D_OUT / 16) * 32, 0, stream>>>(
        agg, deg, h, Wl2Bf, b2, Wr2Bf, out);
}
